// LSTM_7155415515250
// MI455X (gfx1250) — matrix operand order repaired and checked by reading, not compiled, not run on hardware
//
#include <hip/hip_runtime.h>
#include <cstdint>
#include <cstddef>

// ---------------------------------------------------------------------------
// Problem dimensions (fixed by the reference)
// ---------------------------------------------------------------------------
#define N_NODES   2048
#define N_ENV     11
#define N_HID     2048
#define HORIZON   64
#define N_IN      16
#define VEC_LEN   (N_IN * N_NODES)     // 32768 = full input vector length
#define S_STATIC  30720                // static (non-recurrent) prefix of vec
#define GATES     (4 * N_HID)          // 8192
#define TP1       (HORIZON + 1)        // 65

typedef __attribute__((ext_vector_type(16))) _Float16 v16h;
typedef __attribute__((ext_vector_type(8)))  _Float16 v8h;
typedef __attribute__((ext_vector_type(8)))  float    v8f;

// ---------------------------------------------------------------------------
// Helpers
// ---------------------------------------------------------------------------
__device__ __forceinline__ uint16_t f2h_bits(float f) {
  return __builtin_bit_cast(uint16_t, (_Float16)f);   // v_cvt_f16_f32 (RNE)
}
__device__ __forceinline__ uint32_t pack_h2(float x, float y) {
  return (uint32_t)f2h_bits(x) | ((uint32_t)f2h_bits(y) << 16);
}

// 8-MAC fp16xfp32 dot: 8 fp16 weights vs two float4 activations.
// float(h16)*f32+f32 should lower to v_fma_mix_f32 (fp16 extend folded).
__device__ __forceinline__ float dot8h(const v8h w, const float4 a, const float4 b) {
  float s0 = (float)w[0] * a.x;
  float s1 = (float)w[1] * a.y;
  s0 = fmaf((float)w[2], a.z, s0);
  s1 = fmaf((float)w[3], a.w, s1);
  s0 = fmaf((float)w[4], b.x, s0);
  s1 = fmaf((float)w[5], b.y, s1);
  s0 = fmaf((float)w[6], b.z, s0);
  s1 = fmaf((float)w[7], b.w, s1);
  return s0 + s1;
}

__device__ __forceinline__ float wave_red(float v) {
  #pragma unroll
  for (int o = 16; o > 0; o >>= 1) v += __shfl_down(v, o, 32);
  return v;
}
__device__ __forceinline__ float sigmoidf(float x) { return 1.0f / (1.0f + __expf(-x)); }

// WMMA fragment loaders (16-bit layouts, ISA 7.12.2 / 7.12.4).
// A 16x32: lanes 0-15 -> K {0..7,16..23}; lanes 16-31 -> K {8..15,24..31}
__device__ __forceinline__ v16h frag_a(const uint16_t* row, int k0, int hf) {
  const v8h a0 = *(const v8h*)(row + k0 + hf * 8);
  const v8h a1 = *(const v8h*)(row + k0 + 16 + hf * 8);
  v16h a;
  #pragma unroll
  for (int i = 0; i < 8; ++i) { a[i] = a0[i]; a[i + 8] = a1[i]; }
  return a;
}
// B 32x16: lanes 0-15 -> N=lane, K {0..7,16..23}; lanes 16-31 -> K {8..15,24..31}
__device__ __forceinline__ v16h frag_b(const uint16_t* row, int k0, int hf) {
  const v8h b0 = *(const v8h*)(row + k0 + hf * 8);
  const v8h b1 = *(const v8h*)(row + k0 + 16 + hf * 8);
  v16h b;
  #pragma unroll
  for (int i = 0; i < 8; ++i) { b[i] = b0[i]; b[i + 8] = b1[i]; }
  return b;
}

// ---------------------------------------------------------------------------
// fp32 -> fp16 conversions (vectorized: 8 elements / thread / iter)
// ---------------------------------------------------------------------------
__global__ void __launch_bounds__(256)
k_f32_to_f16_v8(const float* __restrict__ src, uint16_t* __restrict__ dst, size_t n8) {
  size_t i = (size_t)blockIdx.x * blockDim.x + threadIdx.x;
  const size_t stride = (size_t)gridDim.x * blockDim.x;
  for (; i < n8; i += stride) {
    const float4 f0 = ((const float4*)src)[2 * i];
    const float4 f1 = ((const float4*)src)[2 * i + 1];
    uint4 o;
    o.x = pack_h2(f0.x, f0.y);
    o.y = pack_h2(f0.z, f0.w);
    o.z = pack_h2(f1.x, f1.y);
    o.w = pack_h2(f1.z, f1.w);
    ((uint4*)dst)[i] = o;
  }
}

// W_in[:, 30720:32768] -> fp16 (2048 x 2048), vectorized gather
__global__ void __launch_bounds__(256)
k_conv_wx(const float* __restrict__ Win, uint16_t* __restrict__ Wx) {
  const size_t idx = (size_t)blockIdx.x * blockDim.x + threadIdx.x;
  const size_t nchunks = (size_t)N_HID * (N_NODES / 8);
  if (idx >= nchunks) return;
  const size_t r = idx / (N_NODES / 8);
  const size_t c = (idx % (N_NODES / 8)) * 8;
  const float* s = Win + r * VEC_LEN + S_STATIC + c;
  const float4 f0 = ((const float4*)s)[0];
  const float4 f1 = ((const float4*)s)[1];
  uint4 o;
  o.x = pack_h2(f0.x, f0.y);
  o.y = pack_h2(f0.z, f0.w);
  o.z = pack_h2(f1.x, f1.y);
  o.w = pack_h2(f1.z, f1.w);
  *(uint4*)(Wx + r * N_NODES + c) = o;
}

// W_in[:, :30720] -> fp16 (2048 x 30720), strided row gather
__global__ void __launch_bounds__(256)
k_conv_a(const float* __restrict__ Win, uint16_t* __restrict__ A) {
  const size_t idx = (size_t)blockIdx.x * blockDim.x + threadIdx.x;
  const size_t nchunks = (size_t)N_HID * (S_STATIC / 8);
  if (idx >= nchunks) return;
  const size_t r = idx / (S_STATIC / 8);
  const size_t c = (idx % (S_STATIC / 8)) * 8;
  const float* s = Win + r * VEC_LEN + c;
  const float4 f0 = ((const float4*)s)[0];
  const float4 f1 = ((const float4*)s)[1];
  uint4 o;
  o.x = pack_h2(f0.x, f0.y);
  o.y = pack_h2(f0.z, f0.w);
  o.z = pack_h2(f1.x, f1.y);
  o.w = pack_h2(f1.z, f1.w);
  *(uint4*)(A + r * S_STATIC + c) = o;
}

// ---------------------------------------------------------------------------
// Pack the static input matrix, transposed: VsT[t][s], fp16, (64 x 30720)
// vec layout: [coords(4096) | env(22528) | dusk(2048) | dawn(2048) | x(dynamic)]
// ---------------------------------------------------------------------------
__global__ void __launch_bounds__(256)
k_pack_vst(const float* __restrict__ coords, const float* __restrict__ env,
           const float* __restrict__ dusk, const float* __restrict__ dawn,
           uint16_t* __restrict__ VsT) {
  const int s = blockIdx.x * blockDim.x + threadIdx.x;
  const int t = blockIdx.y;
  if (s >= S_STATIC) return;
  float v;
  if (s < 4096) {
    v = coords[s];                                         // coords row-major (2048,2)
  } else if (s < 4096 + N_NODES * N_ENV) {
    int k = s - 4096;
    int n = k / N_ENV, e = k % N_ENV;                      // env (2048,11,65), time t+1
    v = env[(size_t)n * (N_ENV * TP1) + e * TP1 + (t + 1)];
  } else if (s < 4096 + N_NODES * N_ENV + N_NODES) {
    int n = s - (4096 + N_NODES * N_ENV);
    v = dusk[(size_t)n * HORIZON + t];                     // local_dusk (2048,64)
  } else {
    int n = s - (4096 + N_NODES * N_ENV + N_NODES);
    v = dawn[(size_t)n * TP1 + (t + 1)];                   // local_dawn (2048,65)
  }
  VsT[(size_t)t * S_STATIC + s] = f2h_bits(v);
}

// ---------------------------------------------------------------------------
// WMMA GEMM (fp16 A path): U(2048x64) = Ah @ VsT^T + b_in
// One wave per 16x16 tile; 960 x v_wmma_f32_16x16x32_f16, unrolled x4.
// ---------------------------------------------------------------------------
__global__ void __launch_bounds__(128)
k_gemm_h(const uint16_t* __restrict__ A, const uint16_t* __restrict__ VsT,
         const float* __restrict__ b_in, float* __restrict__ U) {
  const int wid  = threadIdx.x >> 5;
  const int lane = threadIdx.x & 31;
  const int hf   = lane >> 4;
  const int lm   = lane & 15;
  const int m0 = (blockIdx.x * 4 + wid) * 16;
  const int n0 = blockIdx.y * 16;

  const uint16_t* arow = A   + (size_t)(m0 + lm) * S_STATIC;
  const uint16_t* brow = VsT + (size_t)(n0 + lm) * S_STATIC;

  v8f acc = {};
  #pragma unroll 4
  for (int k0 = 0; k0 < S_STATIC; k0 += 32) {
    const v16h a = frag_a(arow, k0, hf);
    const v16h b = frag_b(brow, k0, hf);
    acc = __builtin_amdgcn_wmma_f32_16x16x32_f16(
        false, a, false, b, (short)0, acc, false, false);
  }
  #pragma unroll
  for (int r = 0; r < 8; ++r) {
    const int m = m0 + r + 8 * hf;
    U[(size_t)m * HORIZON + (n0 + lm)] = acc[r] + b_in[m];
  }
}

// Fallback GEMM (on-the-fly fp32->fp16 A) if workspace is too small for Ah
__global__ void __launch_bounds__(128)
k_gemm_f32(const float* __restrict__ Win, const uint16_t* __restrict__ VsT,
           const float* __restrict__ b_in, float* __restrict__ U) {
  const int wid  = threadIdx.x >> 5;
  const int lane = threadIdx.x & 31;
  const int hf   = lane >> 4;
  const int lm   = lane & 15;
  const int m0 = (blockIdx.x * 4 + wid) * 16;
  const int n0 = blockIdx.y * 16;

  const float*    arow = Win + (size_t)(m0 + lm) * VEC_LEN;
  const uint16_t* brow = VsT + (size_t)(n0 + lm) * S_STATIC;

  v8f acc = {};
  for (int k0 = 0; k0 < S_STATIC; k0 += 32) {
    const int o1 = k0 + hf * 8;
    const int o2 = k0 + 16 + hf * 8;
    const float4 t0 = *(const float4*)(arow + o1);
    const float4 t1 = *(const float4*)(arow + o1 + 4);
    const float4 t2 = *(const float4*)(arow + o2);
    const float4 t3 = *(const float4*)(arow + o2 + 4);
    v16h a;
    a[0]=(_Float16)t0.x; a[1]=(_Float16)t0.y; a[2]=(_Float16)t0.z; a[3]=(_Float16)t0.w;
    a[4]=(_Float16)t1.x; a[5]=(_Float16)t1.y; a[6]=(_Float16)t1.z; a[7]=(_Float16)t1.w;
    a[8]=(_Float16)t2.x; a[9]=(_Float16)t2.y; a[10]=(_Float16)t2.z; a[11]=(_Float16)t2.w;
    a[12]=(_Float16)t3.x; a[13]=(_Float16)t3.y; a[14]=(_Float16)t3.z; a[15]=(_Float16)t3.w;
    const v16h b = frag_b(brow, k0, hf);
    acc = __builtin_amdgcn_wmma_f32_16x16x32_f16(
        false, a, false, b, (short)0, acc, false, false);
  }
  #pragma unroll
  for (int r = 0; r < 8; ++r) {
    const int m = m0 + r + 8 * hf;
    U[(size_t)m * HORIZON + (n0 + lm)] = acc[r] + b_in[m];
  }
}

// ---------------------------------------------------------------------------
// Init: x0 -> x buffer + output column 0; zero h/c
// ---------------------------------------------------------------------------
__global__ void __launch_bounds__(256)
k_init(const float* __restrict__ x, float* __restrict__ xbuf,
       float* __restrict__ h0, float* __restrict__ c0,
       float* __restrict__ h1, float* __restrict__ c1,
       float* __restrict__ dout) {
  const int n = blockIdx.x * blockDim.x + threadIdx.x;
  if (n < N_NODES) {
    const float v = x[(size_t)n * TP1];   // x[:,0]
    xbuf[n] = v;
    dout[(size_t)n * TP1] = v;
    h0[n] = 0.f; c0[n] = 0.f; h1[n] = 0.f; c1[n] = 0.f;
  }
}

// ---------------------------------------------------------------------------
// layer_in[j] = U[j][t] + Wx_fp16[j,:] . x_t   (one wave per row; each lane
// owns 64 contiguous fp16 of the row, loaded as 8x v8h)
// ---------------------------------------------------------------------------
__global__ void __launch_bounds__(256)
k_layerin(const uint16_t* __restrict__ Wx, const float* __restrict__ xin,
          const float* __restrict__ U, float* __restrict__ out, int t) {
  const int row  = blockIdx.x * 8 + (threadIdx.x >> 5);
  const int lane = threadIdx.x & 31;
  const uint16_t* w  = Wx + (size_t)row * N_NODES + lane * 64;
  const float4*   x4 = (const float4*)xin + lane * 16;
  float acc0 = 0.f, acc1 = 0.f;
  #pragma unroll 4
  for (int i = 0; i < 8; ++i) {
    const v8h wv = *(const v8h*)(w + i * 8);
    const float d = dot8h(wv, x4[2 * i], x4[2 * i + 1]);
    if (i & 1) acc1 += d; else acc0 += d;
  }
  float acc = wave_red(acc0 + acc1);
  if (lane == 0) out[row] = acc + U[(size_t)row * HORIZON + t];
}

// ---------------------------------------------------------------------------
// Fused LSTM cell: one wave per hidden unit j; 8 weight streams (4 gates x ih/hh)
// ---------------------------------------------------------------------------
__global__ void __launch_bounds__(256)
k_cell(const uint16_t* __restrict__ Wih, const uint16_t* __restrict__ Whh,
       const float* __restrict__ bih, const float* __restrict__ bhh,
       const float* __restrict__ vin, const float* __restrict__ h_old,
       const float* __restrict__ c_old,
       float* __restrict__ h_new, float* __restrict__ c_new) {
  const int j    = blockIdx.x * 8 + (threadIdx.x >> 5);
  const int lane = threadIdx.x & 31;
  const size_t GS = (size_t)N_HID * N_HID;   // gate stride, fp16 elements
  const uint16_t* Wi = Wih + (size_t)j * N_HID + lane * 64;
  const uint16_t* Wh = Whh + (size_t)j * N_HID + lane * 64;
  const float4* in4 = (const float4*)vin + lane * 16;
  const float4* h4  = (const float4*)h_old + lane * 16;

  float a0 = 0.f, a1 = 0.f, a2 = 0.f, a3 = 0.f;
  #pragma unroll 2
  for (int i = 0; i < 8; ++i) {
    const float4 iA = in4[2 * i], iB = in4[2 * i + 1];
    const float4 hA = h4[2 * i],  hB = h4[2 * i + 1];
    a0 += dot8h(*(const v8h*)(Wi + i * 8),          iA, iB);
    a0 += dot8h(*(const v8h*)(Wh + i * 8),          hA, hB);
    a1 += dot8h(*(const v8h*)(Wi + GS + i * 8),     iA, iB);
    a1 += dot8h(*(const v8h*)(Wh + GS + i * 8),     hA, hB);
    a2 += dot8h(*(const v8h*)(Wi + 2 * GS + i * 8), iA, iB);
    a2 += dot8h(*(const v8h*)(Wh + 2 * GS + i * 8), hA, hB);
    a3 += dot8h(*(const v8h*)(Wi + 3 * GS + i * 8), iA, iB);
    a3 += dot8h(*(const v8h*)(Wh + 3 * GS + i * 8), hA, hB);
  }
  a0 = wave_red(a0); a1 = wave_red(a1); a2 = wave_red(a2); a3 = wave_red(a3);

  if (lane == 0) {
    const float gi = a0 + bih[j]             + bhh[j];
    const float gf = a1 + bih[j + N_HID]     + bhh[j + N_HID];
    const float gg = a2 + bih[j + 2 * N_HID] + bhh[j + 2 * N_HID];
    const float go = a3 + bih[j + 3 * N_HID] + bhh[j + 3 * N_HID];
    const float cn = sigmoidf(gf) * c_old[j] + sigmoidf(gi) * tanhf(gg);
    c_new[j] = cn;
    h_new[j] = sigmoidf(go) * tanhf(cn);
  }
}

// ---------------------------------------------------------------------------
// x_{t+1}[n] = x_t[n] + tanh(W_out[n,:] . h + b_out[n]); writes output column
// ---------------------------------------------------------------------------
__global__ void __launch_bounds__(256)
k_out(const uint16_t* __restrict__ Wout, const float* __restrict__ h,
      const float* __restrict__ b_out, const float* __restrict__ x_old,
      float* __restrict__ x_new, float* __restrict__ dout, int t) {
  const int row  = blockIdx.x * 8 + (threadIdx.x >> 5);
  const int lane = threadIdx.x & 31;
  const uint16_t* w  = Wout + (size_t)row * N_HID + lane * 64;
  const float4*   h4 = (const float4*)h + lane * 16;
  float acc0 = 0.f, acc1 = 0.f;
  #pragma unroll 4
  for (int i = 0; i < 8; ++i) {
    const v8h wv = *(const v8h*)(w + i * 8);
    const float d = dot8h(wv, h4[2 * i], h4[2 * i + 1]);
    if (i & 1) acc1 += d; else acc0 += d;
  }
  float acc = wave_red(acc0 + acc1);
  if (lane == 0) {
    const float v = x_old[row] + tanhf(acc + b_out[row]);
    x_new[row] = v;
    dout[(size_t)row * TP1 + (t + 1)] = v;
  }
}

// ---------------------------------------------------------------------------
// Host-side orchestration
// ---------------------------------------------------------------------------
extern "C" void kernel_launch(void* const* d_in, const int* in_sizes, int n_in,
                              void* d_out, int out_size, void* d_ws, size_t ws_size,
                              hipStream_t stream) {
  (void)in_sizes; (void)n_in; (void)out_size;

  const float* x      = (const float*)d_in[0];
  const float* coords = (const float*)d_in[1];
  const float* env    = (const float*)d_in[2];
  const float* dusk   = (const float*)d_in[3];
  const float* dawn   = (const float*)d_in[4];
  const float* W_in   = (const float*)d_in[5];
  const float* b_in   = (const float*)d_in[6];
  const float* W_ih   = (const float*)d_in[7];
  const float* W_hh   = (const float*)d_in[8];
  const float* b_ih   = (const float*)d_in[9];
  const float* b_hh   = (const float*)d_in[10];
  const float* W_out  = (const float*)d_in[11];
  const float* b_out  = (const float*)d_in[12];
  float* out = (float*)d_out;

  // ---- workspace carve-up (256B aligned) ----
  char* ws = (char*)d_ws;
  size_t off = 0;
  auto alloc = [&](size_t bytes) -> void* {
    void* p = ws + off;
    off += (bytes + 255) & ~(size_t)255;
    return p;
  };
  uint16_t* WihH  = (uint16_t*)alloc((size_t)2 * GATES * N_HID * 2);   // 64 MB
  uint16_t* WhhH  = (uint16_t*)alloc((size_t)2 * GATES * N_HID * 2);   // 64 MB
  uint16_t* WxH   = (uint16_t*)alloc((size_t)N_HID * N_NODES * 2);     //  8 MB
  uint16_t* WoutH = (uint16_t*)alloc((size_t)N_NODES * N_HID * 2);     //  8 MB
  uint16_t* VsT   = (uint16_t*)alloc((size_t)HORIZON * S_STATIC * 2);  // 3.75 MB
  float*    U     = (float*)alloc((size_t)N_HID * HORIZON * 4);        // 512 KB
  float*    lin   = (float*)alloc((size_t)N_HID * 4);
  float* xb[2]  = { (float*)alloc(N_NODES * 4), (float*)alloc(N_NODES * 4) };
  float* h0b[2] = { (float*)alloc(N_HID * 4),  (float*)alloc(N_HID * 4) };
  float* h1b[2] = { (float*)alloc(N_HID * 4),  (float*)alloc(N_HID * 4) };
  float* c0b[2] = { (float*)alloc(N_HID * 4),  (float*)alloc(N_HID * 4) };
  float* c1b[2] = { (float*)alloc(N_HID * 4),  (float*)alloc(N_HID * 4) };
  // Optional fp16 copy of the static W_in slab (120 MB) -- only if ws allows.
  const size_t need_a = (size_t)N_HID * S_STATIC * 2;
  uint16_t* Ah = (ws_size >= off + need_a) ? (uint16_t*)alloc(need_a) : nullptr;

  // ---- one-time per call: fp16 weight conversion (step weights fit in L2) ----
  {
    const size_t n8w = (size_t)2 * GATES * N_HID / 8;
    k_f32_to_f16_v8<<<(int)((n8w + 255) / 256), 256, 0, stream>>>(W_ih, WihH, n8w);
    k_f32_to_f16_v8<<<(int)((n8w + 255) / 256), 256, 0, stream>>>(W_hh, WhhH, n8w);
    const size_t n8o = (size_t)N_NODES * N_HID / 8;
    k_f32_to_f16_v8<<<(int)((n8o + 255) / 256), 256, 0, stream>>>(W_out, WoutH, n8o);
    const size_t nwx = (size_t)N_HID * (N_NODES / 8);
    k_conv_wx<<<(int)((nwx + 255) / 256), 256, 0, stream>>>(W_in, WxH);
  }

  // ---- pack static inputs, then the hoisted WMMA GEMM (W_in read ONCE) ----
  k_pack_vst<<<dim3((S_STATIC + 255) / 256, HORIZON), 256, 0, stream>>>(
      coords, env, dusk, dawn, VsT);
  if (Ah) {
    const size_t na = (size_t)N_HID * (S_STATIC / 8);
    k_conv_a<<<(int)((na + 255) / 256), 256, 0, stream>>>(W_in, Ah);
    k_gemm_h<<<dim3(N_HID / 64, HORIZON / 16), 128, 0, stream>>>(Ah, VsT, b_in, U);
  } else {
    k_gemm_f32<<<dim3(N_HID / 64, HORIZON / 16), 128, 0, stream>>>(W_in, VsT, b_in, U);
  }

  // ---- recurrence: 64 steps x 4 stream-ordered kernels ----
  k_init<<<(N_NODES + 255) / 256, 256, 0, stream>>>(
      x, xb[0], h0b[0], c0b[0], h1b[0], c1b[0], out);

  const size_t L1OFF = (size_t)GATES * N_HID;   // fp16 elements per layer
  int cur = 0;
  for (int t = 0; t < HORIZON; ++t) {
    const int nxt = cur ^ 1;
    k_layerin<<<N_HID / 8, 256, 0, stream>>>(WxH, xb[cur], U, lin, t);
    k_cell<<<N_HID / 8, 256, 0, stream>>>(
        WihH, WhhH, b_ih, b_hh,
        lin, h0b[cur], c0b[cur], h0b[nxt], c0b[nxt]);
    k_cell<<<N_HID / 8, 256, 0, stream>>>(
        WihH + L1OFF, WhhH + L1OFF, b_ih + GATES, b_hh + GATES,
        h0b[nxt], h1b[cur], c1b[cur], h1b[nxt], c1b[nxt]);
    k_out<<<N_NODES / 8, 256, 0, stream>>>(
        WoutH, h1b[nxt], b_out, xb[cur], xb[nxt], out, t);
    cur = nxt;
  }
}
